// DecodingLoss_83597243449419
// MI455X (gfx1250) — matrix operand order repaired and checked by reading, not compiled, not run on hardware
//
#include <hip/hip_runtime.h>
#include <math.h>

typedef __attribute__((ext_vector_type(16))) _Float16 v16h;
typedef __attribute__((ext_vector_type(8)))  _Float16 v8h;
typedef __attribute__((ext_vector_type(8)))  float    v8f;

#define WIN 4096
#define NT  256
#define PI_F 3.14159265358979f

// ---- WMMA helpers (CDNA5 gfx1250, wave32) -------------------------------

__device__ __forceinline__ v8f wmma_f16(v16h a, v16h b, v8f c) {
    // D = A(16x32 f16) x B(32x16 f16) + C(16x16 f32)
    return __builtin_amdgcn_wmma_f32_16x16x32_f16(
        /*neg_a=*/false, a, /*neg_b=*/false, b,
        /*c_mod=*/(short)0, c, /*reuse_a=*/false, /*reuse_b=*/false);
}

// A fragment: 16x32 tile of row-major [64x64] f16 LDS matrix.
// Lane L: m = L%16, s = L/16; elem j<8 -> K = 8s+j ; j>=8 -> K = 16+8s+(j-8)
__device__ __forceinline__ v16h loadA(const _Float16* lds, int row0, int col0, int lane) {
    int m = lane & 15, s = lane >> 4;
    const _Float16* p = lds + (row0 + m) * 64 + col0 + 8 * s;
    v8h lo = *(const v8h*)(p);
    v8h hi = *(const v8h*)(p + 16);
    v16h a;
#pragma unroll
    for (int j = 0; j < 8; ++j) { a[j] = lo[j]; a[8 + j] = hi[j]; }
    return a;
}

// B fragment: 32x16 tile B[k][n]; storage is TRANSPOSED (ldsT[n*64 + k]),
// which also works directly for symmetric matrices (DFT cos/sin).
// Lane L: n = L%16, s = L/16; elem j<8 -> K = 8s+j ; j>=8 -> K = 16+8s+(j-8)
__device__ __forceinline__ v16h loadBT(const _Float16* ldsT, int row0, int col0, int lane) {
    int n = lane & 15, kb = (lane >> 4) * 8;
    const _Float16* p = ldsT + (col0 + n) * 64 + row0 + kb;
    v8h lo = *(const v8h*)(p);
    v8h hi = *(const v8h*)(p + 16);
    v16h b;
#pragma unroll
    for (int j = 0; j < 8; ++j) { b[j] = lo[j]; b[8 + j] = hi[j]; }
    return b;
}

__device__ __forceinline__ v16h negh(v16h x) {
    v16h r;
#pragma unroll
    for (int j = 0; j < 16; ++j) r[j] = -x[j];
    return r;
}

// ---- init: cosine table + precomputed f16 DFT64 matrices; zero errs -----

__global__ void init_tab_kernel(float* __restrict__ gtab,
                                _Float16* __restrict__ gC16,
                                _Float16* __restrict__ gS16,
                                float* __restrict__ errs, int B) {
    int i = blockIdx.x * blockDim.x + threadIdx.x;
    if (i < WIN) {
        gtab[i] = __cosf((float)i * (2.0f * PI_F / (float)WIN));
        int r = i >> 6, c = i & 63;
        float ang = (float)((r * c) & 63) * (2.0f * PI_F / 64.0f);
        float sv, cv;
        __sincosf(ang, &sv, &cv);
        gC16[i] = (_Float16)cv;      // cos DFT64 (symmetric), fragment layout-ready
        gS16[i] = (_Float16)sv;      // sin DFT64 (symmetric)
    }
    if (i < B) errs[i] = 0.0f;
}

// ---- per-window kernel: radix-64x64 FFT via WMMA, cepstrum at 8 lags ----
// All trig = integer-phase lookups: cos(2*pi*p/4096) = ctab[p&4095],
// sin(2*pi*p/4096) = ctab[(p+3072)&4095].
// Imag parts carry a flipped sign (Zi' = -Zi) so stage A needs no negation
// and stage C needs only one negated B fragment; |X| is sign-agnostic.

__global__ __launch_bounds__(NT)
void decode_win_kernel(const float* __restrict__ audio,
                       const int* __restrict__ symbols,
                       const float* __restrict__ gtab,
                       const _Float16* __restrict__ gC16,
                       const _Float16* __restrict__ gS16,
                       float* __restrict__ errs,
                       int num_wins)
{
    __shared__ float ctab[WIN];                       // 16 KB cosine table
    __shared__ __align__(16) _Float16 sC [64 * 64];   // cos DFT64 (symmetric)
    __shared__ __align__(16) _Float16 sS [64 * 64];   // sin DFT64 (symmetric)
    __shared__ __align__(16) _Float16 sXT[64 * 64];   // Xmat[n1][n2] transposed
    __shared__ __align__(16) _Float16 sZr[64 * 64];   // twiddled stage-A (real)
    __shared__ __align__(16) _Float16 sZi[64 * 64];   // twiddled stage-A (-imag)
    __shared__ float T0[64];                          // sum_k2 L[k1,k2]
    __shared__ float T1[64];                          // sum_k2 (-1)^k2 L[k1,k2]
    __shared__ float red[8];

    const int tid  = threadIdx.x;
    const int lane = tid & 31;
    const int wave = tid >> 5;
    const int win  = blockIdx.x;                 // win = b*num_wins + w
    const int b    = win / num_wins;
    const size_t base = (size_t)win * WIN;

    // Phase 1: bulk-copy tables (uint4), load window as f16 (float4 loads,
    // transposed scatter into LDS), init reducers.
    {
        const uint4* gC4 = (const uint4*)gC16;
        const uint4* gS4 = (const uint4*)gS16;
        const uint4* gt4 = (const uint4*)gtab;
        uint4* sC4 = (uint4*)sC;
        uint4* sS4 = (uint4*)sS;
        uint4* ct4 = (uint4*)ctab;
        sC4[tid]       = gC4[tid];               // 2 x 8KB f16 matrices
        sC4[tid + 256] = gC4[tid + 256];
        sS4[tid]       = gS4[tid];
        sS4[tid + 256] = gS4[tid + 256];
#pragma unroll
        for (int i = 0; i < 4; ++i)              // 16KB f32 cosine table
            ct4[tid + i * NT] = gt4[tid + i * NT];

        const float4* a4 = (const float4*)(audio + base);
#pragma unroll
        for (int i = 0; i < 4; ++i) {
            int chunk = tid + i * NT;            // 4 floats per chunk
            float4 v = a4[chunk];
            int r  = chunk >> 4;                 // n1
            int c0 = (chunk & 15) * 4;           // n2 base
            sXT[(c0 + 0) * 64 + r] = (_Float16)v.x;
            sXT[(c0 + 1) * 64 + r] = (_Float16)v.y;
            sXT[(c0 + 2) * 64 + r] = (_Float16)v.z;
            sXT[(c0 + 3) * 64 + r] = (_Float16)v.w;
        }
    }
    if (tid < 64) { T0[tid] = 0.0f; T1[tid] = 0.0f; }
    __syncthreads();

    // Each wave owns two 16x16 output tiles sharing one tile-column.
    const int tr0 = (wave >> 2) * 16;
    const int tr1 = tr0 + 32;
    const int tc  = (wave & 3) * 16;

    // ---- Stage A: Yr = C*X ; Yi' = S*X (= -Im) --------------------------
    v8f yr0 = {}, yi0 = {}, yr1 = {}, yi1 = {};
#pragma unroll
    for (int kc = 0; kc < 64; kc += 32) {
        v16h bX  = loadBT(sXT, kc, tc, lane);          // shared by both tiles
        v16h aC0 = loadA(sC, tr0, kc, lane);
        v16h aS0 = loadA(sS, tr0, kc, lane);
        v16h aC1 = loadA(sC, tr1, kc, lane);
        v16h aS1 = loadA(sS, tr1, kc, lane);
        yr0 = wmma_f16(aC0, bX, yr0);
        yi0 = wmma_f16(aS0, bX, yi0);
        yr1 = wmma_f16(aC1, bX, yr1);
        yi1 = wmma_f16(aS1, bX, yi1);
    }
    // Twiddle by e^{-2*pi*i*k1*n2/4096}: Zr = yr*cv - yi'*sv ; Zi' = yi'*cv + yr*sv
    {
        int n2  = tc + (lane & 15);
        int mb0 = tr0 + 8 * (lane >> 4);
        int mb1 = tr1 + 8 * (lane >> 4);
#pragma unroll
        for (int r = 0; r < 8; ++r) {
            int k1 = mb0 + r;                    // k1*n2 < 4096
            int p  = k1 * n2;
            float cv = ctab[p], sv = ctab[(p + 3072) & 4095];
            sZr[k1 * 64 + n2] = (_Float16)(yr0[r] * cv - yi0[r] * sv);
            sZi[k1 * 64 + n2] = (_Float16)(yi0[r] * cv + yr0[r] * sv);
            int k1b = mb1 + r;
            int pb  = k1b * n2;
            float cvb = ctab[pb], svb = ctab[(pb + 3072) & 4095];
            sZr[k1b * 64 + n2] = (_Float16)(yr1[r] * cvb - yi1[r] * svb);
            sZi[k1b * 64 + n2] = (_Float16)(yi1[r] * cvb + yr1[r] * svb);
        }
    }
    __syncthreads();

    // ---- Stage C: Xr = Zr*C - Zi'*S ; Xi' = Zi'*C + Zr*S ----------------
    v8f xr0 = {}, xi0 = {}, xr1 = {}, xi1 = {};
#pragma unroll
    for (int kc = 0; kc < 64; kc += 32) {
        v16h bC  = loadBT(sC, kc, tc, lane);           // shared by both tiles
        v16h bS  = loadBT(sS, kc, tc, lane);
        v16h bSn = negh(bS);
        v16h a0r = loadA(sZr, tr0, kc, lane);
        v16h a0i = loadA(sZi, tr0, kc, lane);
        xr0 = wmma_f16(a0r, bC,  xr0);
        xr0 = wmma_f16(a0i, bSn, xr0);
        xi0 = wmma_f16(a0i, bC,  xi0);
        xi0 = wmma_f16(a0r, bS,  xi0);
        v16h a1r = loadA(sZr, tr1, kc, lane);
        v16h a1i = loadA(sZi, tr1, kc, lane);
        xr1 = wmma_f16(a1r, bC,  xr1);
        xr1 = wmma_f16(a1i, bSn, xr1);
        xi1 = wmma_f16(a1i, bC,  xi1);
        xi1 = wmma_f16(a1r, bS,  xi1);
    }

    // Log-magnitude, then separable cepstrum reduction:
    // cos(2*pi*k*d/4096) = (-1)^{k2*(i+2)} * cos(2*pi*k1*(i+2)/128), d=32*(i+2)
    // -> even lags use T0[k1] = sum_k2 L ; odd lags use T1[k1] = sum_k2 (-1)^k2 L.
    {
        int k2  = tc + (lane & 15);
        float sgn = (k2 & 1) ? -1.0f : 1.0f;
        int mb0 = tr0 + 8 * (lane >> 4);
        int mb1 = tr1 + 8 * (lane >> 4);
#pragma unroll
        for (int r = 0; r < 8; ++r) {
            float L0 = __logf(sqrtf(xr0[r] * xr0[r] + xi0[r] * xi0[r]) + 1e-5f);
            float L1 = __logf(sqrtf(xr1[r] * xr1[r] + xi1[r] * xi1[r]) + 1e-5f);
            float a0 = L0, b0 = sgn * L0;
            float a1 = L1, b1 = sgn * L1;
#pragma unroll
            for (int off = 1; off <= 8; off <<= 1) {   // reduce over 16 k2-lanes
                a0 += __shfl_xor(a0, off, 32);
                b0 += __shfl_xor(b0, off, 32);
                a1 += __shfl_xor(a1, off, 32);
                b1 += __shfl_xor(b1, off, 32);
            }
            if ((lane & 15) == 0) {
                atomicAdd(&T0[mb0 + r], a0);
                atomicAdd(&T1[mb0 + r], b0);
                atomicAdd(&T0[mb1 + r], a1);
                atomicAdd(&T1[mb1 + r], b1);
            }
        }
    }
    __syncthreads();

    // Final 8x64 dot: cep[i] = sum_k1 cos(2*pi*k1*(i+2)/128) * T_{parity}[k1]
    if (tid < 8) {
        int step = (tid + 2) << 5;                     // phase step = (i+2)*32
        const float* T = ((tid & 1) == 0) ? T0 : T1;   // (i+2) even -> T0
        float a = 0.0f;
        for (int k1 = 0; k1 < 64; ++k1)
            a += T[k1] * ctab[(k1 * step) & 4095];
        red[tid] = a;
    }
    __syncthreads();

    if (tid == 0) {
        float vals[8], m = -3.0e38f;
#pragma unroll
        for (int d = 0; d < 8; ++d) {
            vals[d] = red[d] * (1.0f / 4096.0f);
            m = fmaxf(m, vals[d]);
        }
        float se = 0.f, sw = 0.f;
#pragma unroll
        for (int d = 0; d < 8; ++d) {
            float e = __expf((vals[d] - m) * 1.0e10f);  // stable softargmax
            se += e;
            sw += e * (float)d;
        }
        float val  = sw / se;
        float sym  = (float)symbols[win];
        float loss = fminf(fabsf(val - sym), 1.0f);
        atomicAdd(&errs[b], loss);
    }
}

// ---- post kernel --------------------------------------------------------

__global__ void finalize_kernel(const float* __restrict__ errs,
                                const int* __restrict__ nn_i,
                                const int* __restrict__ nr_i,
                                float* __restrict__ out,
                                int B, int num_wins) {
    if (threadIdx.x == 0 && blockIdx.x == 0) {
        float tot = 0.f, totred = 0.f, snn = 0.f, snr = 0.f;
        for (int b = 0; b < B; ++b) {
            float e  = errs[b];
            float nn = (float)nn_i[b];
            float nr = (float)nr_i[b];
            float diff = nr - nn;
            totred += (diff == 0.0f) ? 1.0f : diff / (nr - e);
            tot += e; snn += nn; snr += nr;
        }
        float denom = (float)B * (float)num_wins;
        out[0] = tot / denom;          // symbol_err_rate
        out[1] = tot;                  // tot_symbol_errs
        out[2] = totred / (float)B;    // avg_err_reduction
        out[3] = snn / denom;          // gt_ser_no_reverb
        out[4] = snr / denom;          // gt_ser_reverb
    }
}

extern "C" void kernel_launch(void* const* d_in, const int* in_sizes, int n_in,
                              void* d_out, int out_size, void* d_ws, size_t ws_size,
                              hipStream_t stream) {
    const float* audio   = (const float*)d_in[0];
    const int*   symbols = (const int*)d_in[1];
    const int*   nn      = (const int*)d_in[2];
    const int*   nr      = (const int*)d_in[3];
    float* out = (float*)d_out;

    // Workspace layout: [errs: B f32][gtab: 4096 f32][gC16,gS16: 4096 f16 each]
    float*    errs = (float*)d_ws;
    float*    gtab = (float*)((char*)d_ws + 4096);
    _Float16* gC16 = (_Float16*)((char*)d_ws + 4096 + 16384);
    _Float16* gS16 = (_Float16*)((char*)d_ws + 4096 + 16384 + 8192);

    const int B        = in_sizes[2];
    const int num_wins = in_sizes[1] / B;
    const int nblocks  = B * num_wins;

    init_tab_kernel<<<(WIN + NT - 1) / NT, NT, 0, stream>>>(gtab, gC16, gS16, errs, B);
    decode_win_kernel<<<nblocks, NT, 0, stream>>>(audio, symbols, gtab, gC16, gS16,
                                                  errs, num_wins);
    finalize_kernel<<<1, 64, 0, stream>>>(errs, nn, nr, out, B, num_wins);
}
